// Model_18726057411221
// MI455X (gfx1250) — matrix operand order repaired and checked by reading, not compiled, not run on hardware
//
#include <hip/hip_runtime.h>
#include <hip/hip_bf16.h>

// CDNA5 / gfx1250: wave32, WMMA 16x16x32 bf16 (f32 accumulate).
typedef __attribute__((ext_vector_type(16))) __bf16 v16bf;
typedef __attribute__((ext_vector_type(8)))  float  v8f;

#define N_NODES   50000
#define N_EDGES   800000
#define IN_C      128
#define HID       64
#define N_GRAPHS  256
#define N_CLASSES 10
#define MT_STRIP  8                  // m-tiles per wave in the GEMM
#define MT_TILES  (N_NODES / 16)     // 3125 (exact)
#define NT_TILES  (HID / 16)         // 4

// ---------------------------------------------------------------- utilities
__global__ void k_zero(float* __restrict__ p, int n) {
    int i = blockIdx.x * blockDim.x + threadIdx.x;
    if (i < n) p[i] = 0.f;
}

// Pack W[K,N] (f32 row-major) into WMMA B-fragment order:
// dst[(((nt*KT + kt)*32 + lane)*16 + e)] = bf16( W[(kt*32 + 16*(lane>=16) + e)*N + nt*16 + (lane&15)] )
__global__ void k_pack_w(__bf16* __restrict__ dst, const float* __restrict__ src,
                         int K, int N) {
    int i = blockIdx.x * blockDim.x + threadIdx.x;
    if (i >= K * N) return;
    int e    = i & 15;
    int lane = (i >> 4) & 31;
    int rest = i >> 9;
    int KT   = K >> 5;
    int kt   = rest % KT;
    int nt   = rest / KT;
    int k = (kt << 5) + ((lane >> 4) << 4) + e;
    int n = (nt << 4) + (lane & 15);
    dst[i] = (__bf16)src[(size_t)k * N + n];
}

// ---------------------------------------------------------------- scatter-add over edges
__global__ void k_scatter(const float* __restrict__ H, int ldH,
                          const int* __restrict__ src, const int* __restrict__ dst,
                          float* __restrict__ agg, int C) {
    int e = blockIdx.x * blockDim.x + threadIdx.x;
    if (e >= N_EDGES) return;
    int c = blockIdx.y << 2;
    int s = src[e], d = dst[e];
    const float4 v = *(const float4*)(H + (size_t)s * ldH + c);
    float* o = agg + (size_t)d * C + c;
    atomicAdd(o + 0, v.x); atomicAdd(o + 1, v.y);
    atomicAdd(o + 2, v.z); atomicAdd(o + 3, v.w);
}

// out[m*C+c] = a[m*lda+c] + agg[m*C+c]
__global__ void k_add(float* __restrict__ out, const float* __restrict__ a, int lda,
                      const float* __restrict__ agg, int C, int n) {
    int i = blockIdx.x * blockDim.x + threadIdx.x;
    if (i >= n) return;
    int m = i / C, c = i - m * C;
    out[i] = a[(size_t)m * lda + c] + agg[i];
}

// ---------------------------------------------------------------- WMMA GEMM
// Out[M,64](ld=LDO) = act( A[M,K](ldA, f32) @ W(packed bf16 fragments) + bias )
// One wave: one 16-wide n-tile x MT_STRIP m-tiles. B fragments live in registers.
template<int KT, int LDO, bool RELU>
__global__ void k_gemm_wmma(const float* __restrict__ A, int ldA,
                            const __bf16* __restrict__ Wp,
                            const float* __restrict__ bias,
                            float* __restrict__ Out) {
    const int lane   = threadIdx.x & 31;
    const int wave   = blockIdx.x * (blockDim.x >> 5) + (threadIdx.x >> 5);
    const int nt     = wave & (NT_TILES - 1);
    const int mstrip = wave >> 2;
    const int mt0    = mstrip * MT_STRIP;
    if (mt0 >= MT_TILES) return;           // wave-uniform: EXEC all-ones for WMMA

    const int r  = lane & 15;
    const int hi = lane >> 4;

    // Preload all B fragments for this n-tile (2x global_load_b128 each).
    v16bf bfrag[KT];
    #pragma unroll
    for (int kt = 0; kt < KT; ++kt)
        bfrag[kt] = *(const v16bf*)(Wp + ((((size_t)nt * KT + kt) << 5) + lane) * 16);

    const int col  = (nt << 4) + r;
    const float bv = bias[col];

    const int iters = ((mt0 + MT_STRIP <= MT_TILES) ? MT_STRIP : (MT_TILES - mt0));
    const float* Ab = A + (size_t)(mt0 * 16 + r) * ldA + (hi << 4);
    float*       Ob = Out + (size_t)(mt0 * 16 + hi * 8) * LDO + col;

    for (int it = 0; it < iters; ++it) {
        v8f acc = {0.f, 0.f, 0.f, 0.f, 0.f, 0.f, 0.f, 0.f};
        #pragma unroll
        for (int kt = 0; kt < KT; ++kt) {
            // A fragment (16x32 bf16): lane<16 -> K=kk+[0..15]; lane>=16 -> K=kk+[16..31]
            const float* Ap = Ab + (kt << 5);
            float4 a0 = *(const float4*)(Ap);
            float4 a1 = *(const float4*)(Ap + 4);
            float4 a2 = *(const float4*)(Ap + 8);
            float4 a3 = *(const float4*)(Ap + 12);
            v16bf af;
            af[0]  = (__bf16)a0.x; af[1]  = (__bf16)a0.y; af[2]  = (__bf16)a0.z; af[3]  = (__bf16)a0.w;
            af[4]  = (__bf16)a1.x; af[5]  = (__bf16)a1.y; af[6]  = (__bf16)a1.z; af[7]  = (__bf16)a1.w;
            af[8]  = (__bf16)a2.x; af[9]  = (__bf16)a2.y; af[10] = (__bf16)a2.z; af[11] = (__bf16)a2.w;
            af[12] = (__bf16)a3.x; af[13] = (__bf16)a3.y; af[14] = (__bf16)a3.z; af[15] = (__bf16)a3.w;

            acc = __builtin_amdgcn_wmma_f32_16x16x32_bf16(
                false, af, false, bfrag[kt], (short)0, acc, false, false);
        }
        // D layout: element q -> row = m0 + q + 8*hi, col  (immediate-offset stores)
        #pragma unroll
        for (int q = 0; q < 8; ++q) {
            float v = acc[q] + bv;
            if (RELU) v = fmaxf(v, 0.f);
            Ob[q * LDO] = v;
        }
        Ab += (size_t)16 * ldA;
        Ob += (size_t)16 * LDO;
    }
}

// ---------------------------------------------------------------- pooling + head
__global__ void k_pool(const float* __restrict__ H, const int* __restrict__ batch,
                       float* __restrict__ g) {
    int n = blockIdx.x * blockDim.x + threadIdx.x;
    if (n >= N_NODES) return;
    int c = blockIdx.y << 2;
    int b = batch[n];
    const float4 v = *(const float4*)(H + (size_t)n * HID + c);
    float* o = g + (size_t)b * HID + c;
    atomicAdd(o + 0, v.x); atomicAdd(o + 1, v.y);
    atomicAdd(o + 2, v.z); atomicAdd(o + 3, v.w);
}

__global__ void k_head(const float* __restrict__ g,
                       const float* __restrict__ c1w, const float* __restrict__ c1b,
                       const float* __restrict__ bng, const float* __restrict__ bnb,
                       const float* __restrict__ bnrm, const float* __restrict__ bnrv,
                       const float* __restrict__ c2w, const float* __restrict__ c2b,
                       float* __restrict__ out) {
    __shared__ float row[HID];
    __shared__ float mid[HID];
    const int gi = blockIdx.x, t = threadIdx.x;
    row[t] = g[(size_t)gi * HID + t];
    __syncthreads();
    float acc = c1b[t];
    #pragma unroll 8
    for (int k = 0; k < HID; ++k) acc += row[k] * c1w[k * HID + t];
    acc = (acc - bnrm[t]) * rsqrtf(bnrv[t] + 1e-5f) * bng[t] + bnb[t];
    mid[t] = fmaxf(acc, 0.f);
    __syncthreads();
    if (t < N_CLASSES) {
        float o = c2b[t];
        #pragma unroll 8
        for (int k = 0; k < HID; ++k) o += mid[k] * c2w[k * N_CLASSES + t];
        out[(size_t)gi * N_CLASSES + t] = o;
    }
}

// ---------------------------------------------------------------- launch
extern "C" void kernel_launch(void* const* d_in, const int* in_sizes, int n_in,
                              void* d_out, int out_size, void* d_ws, size_t ws_size,
                              hipStream_t stream) {
    (void)in_sizes; (void)n_in; (void)out_size; (void)ws_size;

    const float* x     = (const float*)d_in[0];
    const int*   ei    = (const int*)d_in[1];
    const int*   src   = ei;
    const int*   dst   = ei + N_EDGES;
    const int*   batch = (const int*)d_in[2];
    const float* w1[3] = {(const float*)d_in[3],  (const float*)d_in[7],  (const float*)d_in[11]};
    const float* b1[3] = {(const float*)d_in[4],  (const float*)d_in[8],  (const float*)d_in[12]};
    const float* w2[3] = {(const float*)d_in[5],  (const float*)d_in[9],  (const float*)d_in[13]};
    const float* b2[3] = {(const float*)d_in[6],  (const float*)d_in[10], (const float*)d_in[14]};
    const float* jkw   = (const float*)d_in[15];
    const float* jkb   = (const float*)d_in[16];
    const float* c1w   = (const float*)d_in[17];
    const float* c1b   = (const float*)d_in[18];
    const float* bng   = (const float*)d_in[19];
    const float* bnb   = (const float*)d_in[20];
    const float* bnrm  = (const float*)d_in[21];
    const float* bnrv  = (const float*)d_in[22];
    const float* c2w   = (const float*)d_in[23];
    const float* c2b   = (const float*)d_in[24];
    float* out = (float*)d_out;

    // ---- workspace layout (~102.6 MB) ----
    char* ws = (char*)d_ws;
    __bf16* wb = (__bf16*)ws;                       // packed bf16 weights (81920 B)
    size_t off = 128 * 1024;
    float* agg  = (float*)(ws + off); off += (size_t)N_NODES * IN_C * 4;    // 25.6 MB
    float* h128 = (float*)(ws + off); off += (size_t)N_NODES * IN_C * 4;    // 25.6 MB
    float* t    = (float*)(ws + off); off += (size_t)N_NODES * HID * 4;     // 12.8 MB
    float* xs   = (float*)(ws + off); off += (size_t)N_NODES * 3 * HID * 4; // 38.4 MB
    float* g    = (float*)(ws + off); off += (size_t)N_GRAPHS * HID * 4;
    float* h64 = h128;  // reuse (h128 dead after layer 0)
    float* hjk = agg;   // reuse (agg dead after last scatter)

    __bf16* bw1[3]; __bf16* bw2[3]; __bf16* bjk;
    size_t woff = 0;
    bw1[0] = wb + woff; woff += IN_C * HID;
    bw2[0] = wb + woff; woff += HID * HID;
    bw1[1] = wb + woff; woff += HID * HID;
    bw2[1] = wb + woff; woff += HID * HID;
    bw1[2] = wb + woff; woff += HID * HID;
    bw2[2] = wb + woff; woff += HID * HID;
    bjk    = wb + woff; woff += 3 * HID * HID;

    // ---- pack weights into bf16 WMMA fragment order ----
    k_pack_w<<<(IN_C*HID + 255)/256, 256, 0, stream>>>(bw1[0], w1[0], IN_C, HID);
    for (int l = 0; l < 3; ++l) {
        if (l) k_pack_w<<<(HID*HID + 255)/256, 256, 0, stream>>>(bw1[l], w1[l], HID, HID);
        k_pack_w<<<(HID*HID + 255)/256, 256, 0, stream>>>(bw2[l], w2[l], HID, HID);
    }
    k_pack_w<<<(3*HID*HID + 255)/256, 256, 0, stream>>>(bjk, jkw, 3*HID, HID);

    // GEMM launch shape: waves = Nt * ceil(Mt/MT_STRIP), 8 waves/block
    const int nwaves = NT_TILES * ((MT_TILES + MT_STRIP - 1) / MT_STRIP);
    const int GB = (nwaves + 7) / 8;

    // ---- layer 0 (C_in = 128) ----
    k_zero<<<(N_NODES*IN_C + 255)/256, 256, 0, stream>>>(agg, N_NODES*IN_C);
    k_scatter<<<dim3((N_EDGES + 255)/256, IN_C/4), 256, 0, stream>>>(x, IN_C, src, dst, agg, IN_C);
    k_add<<<(N_NODES*IN_C + 255)/256, 256, 0, stream>>>(h128, x, IN_C, agg, IN_C, N_NODES*IN_C);
    k_gemm_wmma<4, HID,   true><<<GB, 256, 0, stream>>>(h128, IN_C, bw1[0], b1[0], t);
    k_gemm_wmma<2, 3*HID, true><<<GB, 256, 0, stream>>>(t, HID, bw2[0], b2[0], xs + 0);

    // ---- layers 1,2 (C = 64), reading previous xs slice (ld = 192) ----
    for (int l = 1; l < 3; ++l) {
        k_zero<<<(N_NODES*HID + 255)/256, 256, 0, stream>>>(agg, N_NODES*HID);
        k_scatter<<<dim3((N_EDGES + 255)/256, HID/4), 256, 0, stream>>>(
            xs + (l-1)*HID, 3*HID, src, dst, agg, HID);
        k_add<<<(N_NODES*HID + 255)/256, 256, 0, stream>>>(
            h64, xs + (l-1)*HID, 3*HID, agg, HID, N_NODES*HID);
        k_gemm_wmma<2, HID,   true><<<GB, 256, 0, stream>>>(h64, HID, bw1[l], b1[l], t);
        k_gemm_wmma<2, 3*HID, true><<<GB, 256, 0, stream>>>(t, HID, bw2[l], b2[l], xs + l*HID);
    }

    // ---- JK concat projection: [50000,192] @ [192,64] + b (no relu) ----
    k_gemm_wmma<6, HID, false><<<GB, 256, 0, stream>>>(xs, 3*HID, bjk, jkb, hjk);

    // ---- global_add_pool ----
    k_zero<<<(N_GRAPHS*HID + 255)/256, 256, 0, stream>>>(g, N_GRAPHS*HID);
    k_pool<<<dim3((N_NODES + 255)/256, HID/4), 256, 0, stream>>>(hjk, batch, g);

    // ---- head: c1 + BN(eval) + relu + c2 ----
    k_head<<<N_GRAPHS, HID, 0, stream>>>(g, c1w, c1b, bng, bnb, bnrm, bnrv, c2w, c2b, out);
}
